// SupConLoss2_33268816675430
// MI455X (gfx1250) — matrix operand order repaired and checked by reading, not compiled, not run on hardware
//
#include <hip/hip_runtime.h>
#include <cstdint>
#include <math.h>

// Problem constants (from reference)
#define B_    4096
#define V_    2
#define D_    128
#define N_    8192            // B*V
#define NTILE 512             // N/16 column tiles
#define MARGIN_F 0.1f
#define INV_T  (1.0f/0.07f)   // TEMPERATURE = 0.07, T/BASE_T = 1.0

typedef __attribute__((ext_vector_type(16))) __bf16 v16bf;
typedef __attribute__((ext_vector_type(8)))  float  v8f;

union V16U { uint4 u[2]; v16bf v; };

// ---- fp32 -> bf16 (RNE) without relying on hip_bf16 internals ----
__device__ __forceinline__ unsigned short f2bf(float x) {
  unsigned int u = __float_as_uint(x);
  u += 0x7fffu + ((u >> 16) & 1u);
  return (unsigned short)(u >> 16);
}
__device__ __forceinline__ float bf2f(unsigned short s) {
  return __uint_as_float(((unsigned int)s) << 16);
}

__global__ void init_kernel(float* o, unsigned int* maxn) { o[0] = 0.0f; maxn[0] = 0u; }

// Phase 0a: features [B,V,D] fp32 -> contrast layout [N,D] bf16 hi + bf16 lo
__global__ __launch_bounds__(256) void convert_kernel(
    const float* __restrict__ f, unsigned short* __restrict__ hi,
    unsigned short* __restrict__ lo) {
  int idx = blockIdx.x * 256 + threadIdx.x;      // over N*D = 1M
  int n = idx >> 7, d = idx & 127;
  int b = n & (B_ - 1), v = n >> 12;             // contrast row n = features[b, v]
  float x = f[((size_t)b * V_ + v) * D_ + d];
  unsigned short h = f2bf(x);
  hi[idx] = h;
  lo[idx] = f2bf(x - bf2f(h));
}

// Phase 0b: per-row squared norms + global max (for the per-row logit upper bound).
__global__ __launch_bounds__(256) void norm_kernel(
    const float* __restrict__ f, float* __restrict__ norm2,
    unsigned int* __restrict__ maxn) {
  int n = blockIdx.x * 256 + threadIdx.x;        // 0..N-1
  int b = n & (B_ - 1), v = n >> 12;
  const float4* p = (const float4*)(f + ((size_t)b * V_ + v) * D_);
  float s = 0.f;
#pragma unroll
  for (int i = 0; i < D_ / 4; ++i) {
    float4 x = p[i];
    s = fmaf(x.x, x.x, s); s = fmaf(x.y, x.y, s);
    s = fmaf(x.z, x.z, s); s = fmaf(x.w, x.w, s);
  }
  norm2[n] = s;
  atomicMax(maxn, __float_as_uint(s));           // s >= 0: bit pattern is monotonic
}

// Phase 1: fused Gram-tile WMMA + shifted-softmax row reductions.
// 128 threads = 4 waves; each wave owns one 16-row anchor block. B tiles
// (16 cols x 128 K, hi+lo bf16 = 8KB) double-buffered in LDS via
// global_load_async_to_lds_b128 (ASYNCcnt), shared across the 4 waves.
__global__ __launch_bounds__(128) void supcon_kernel(
    const unsigned short* __restrict__ hi, const unsigned short* __restrict__ lo,
    const int* __restrict__ labels, const float* __restrict__ norm2,
    const unsigned int* __restrict__ maxn, float* __restrict__ out) {
  __shared__ int s_lab[B_];                                   // 16 KB
  __shared__ __align__(16) unsigned short s_buf[2][4096];     // 2 x 8KB (hi 4KB | lo 4KB)

  const int tid  = threadIdx.x;
  const int lane = tid & 31;
  const int wave = tid >> 5;                // 0..3
  const int rb   = blockIdx.x * 4 + wave;   // row block 0..511
  const int rlo  = lane & 15;               // row-in-block (A) / col-in-tile (B/C)
  const int sel  = lane >> 4;

  for (int i = tid; i < B_; i += 128) s_lab[i] = labels[i];

  const uint32_t ldsb0 = (uint32_t)(uintptr_t)(&s_buf[0][0]);
  const uint32_t ldsb1 = (uint32_t)(uintptr_t)(&s_buf[1][0]);

  // Staging plan: wave covers bytes [wave*2048, +2048) of each 8KB tile.
  // Per-instruction source pointers precomputed once, bumped by 4096 B/tile,
  // so the hot loop is pure async-issues (no per-tile cndmask address math).
  uint32_t    dOff[4];
  const char* srcp[4];
#pragma unroll
  for (int i = 0; i < 4; ++i) {
    uint32_t o = (uint32_t)(wave * 2048 + i * 512 + lane * 16);
    dOff[i] = o;
    srcp[i] = (o < 4096u) ? ((const char*)hi + o) : ((const char*)lo + (o - 4096u));
  }
  auto prefetch = [&](uint32_t base) {
#pragma unroll
    for (int i = 0; i < 4; ++i) {
      uint32_t dst = base + dOff[i];
      asm volatile("global_load_async_to_lds_b128 %0, %1, off"
                   :: "v"(dst), "v"(srcp[i]) : "memory");
      srcp[i] += 4096;                      // next tile's source
    }
  };
  prefetch(ldsb0);                          // tile 0 in flight

  // Persistent A operands (16x128 hi+lo) in ISA A-matrix layout:
  // per 32-K chunk (64B/row): lane<16 -> bytes [0..15]+[32..47]; lane>=16 -> [16..31]+[48..63]
  v16bf ahi[4], alo[4];
  {
    const char* bh = (const char*)hi + ((size_t)(rb * 16 + rlo)) * 256;
    const char* bl = (const char*)lo + ((size_t)(rb * 16 + rlo)) * 256;
#pragma unroll
    for (int kc = 0; kc < 4; ++kc) {
      V16U t;
      t.u[0] = *(const uint4*)(bh + kc * 64 + sel * 16);
      t.u[1] = *(const uint4*)(bh + kc * 64 + 32 + sel * 16);
      ahi[kc] = t.v;
      t.u[0] = *(const uint4*)(bl + kc * 64 + sel * 16);
      t.u[1] = *(const uint4*)(bl + kc * 64 + 32 + sel * 16);
      alo[kc] = t.v;
    }
  }

  // Per-row shift bound: M_i = ||a_i|| * max_j ||a_j|| / T >= max_j logits[i,j].
  // log_prob is shift-invariant, so using M_i instead of the true row max is
  // exact; it removes online-max rescaling (1 exp/element instead of 2 + fmax).
  float bias[8];
  {
    float gmax = sqrtf(__uint_as_float(maxn[0]));
#pragma unroll
    for (int r = 0; r < 8; ++r) {
      int m = (lane < 16) ? r : (8 + r);
      bias[r] = -sqrtf(norm2[rb * 16 + m]) * gmax * INV_T;
    }
  }

  __syncthreads();  // labels staged

  int labA[8];
#pragma unroll
  for (int r = 0; r < 8; ++r) {
    int m = (lane < 16) ? r : (8 + r);
    labA[r] = s_lab[(rb * 16 + m) & (B_ - 1)];
  }
  const int diagR = (lane < 16) ? ((lane < 8) ? lane : -1)
                                : ((lane >= 24) ? (lane - 24) : -1);

  float zz[8], pp[8];
  int   cc[8];
#pragma unroll
  for (int r = 0; r < 8; ++r) { zz[r] = 0.f; pp[r] = 0.f; cc[r] = 0; }

  for (int t = 0; t < NTILE; ++t) {
    asm volatile("s_wait_asynccnt 0x0" ::: "memory");
    __syncthreads();
    if (t + 1 < NTILE) prefetch(((t + 1) & 1) ? ldsb1 : ldsb0);

    const char* buf = (const char*)&s_buf[t & 1][0];
    v8f c = {0.f, 0.f, 0.f, 0.f, 0.f, 0.f, 0.f, 0.f};
#pragma unroll
    for (int kc = 0; kc < 4; ++kc) {
      // B operand (K x 16): lane = N; lanes 0-15 hold K 0..7,16..23, lanes 16-31 hold
      // K 8..15,24..31 -> two 16B chunks per lane per 32-K chunk (matches A).
      const char* rbase = buf + rlo * 256 + kc * 64 + sel * 16;
      V16U bh, bl;
      bh.u[0] = *(const uint4*)(rbase);
      bh.u[1] = *(const uint4*)(rbase + 32);
      bl.u[0] = *(const uint4*)(rbase + 4096);
      bl.u[1] = *(const uint4*)(rbase + 4096 + 32);
      // dot = hi*hi + hi*lo + lo*hi (bf16 split compensation, ~fp32 accuracy)
      c = __builtin_amdgcn_wmma_f32_16x16x32_bf16(false, ahi[kc], false, bh.v,
                                                  (short)0, c, false, false);
      c = __builtin_amdgcn_wmma_f32_16x16x32_bf16(false, ahi[kc], false, bl.v,
                                                  (short)0, c, false, false);
      c = __builtin_amdgcn_wmma_f32_16x16x32_bf16(false, alo[kc], false, bh.v,
                                                  (short)0, c, false, false);
    }

    const int  labJ  = s_lab[(t * 16 + rlo) & (B_ - 1)];
    const bool dtile = (t == rb);
#pragma unroll
    for (int r = 0; r < 8; ++r) {
      bool  pos   = (labA[r] == labJ);
      float logit = fmaf(c[r], INV_T, pos ? (-MARGIN_F * INV_T) : 0.f);
      bool  diag  = dtile && (r == diagR);
      float e     = diag ? 0.f : __expf(logit + bias[r]);   // single exp/element
      zz[r] += e;
      if (pos && !diag) { pp[r] += logit; cc[r] += 1; }
    }
  }

  // merge partial row sums across the 16 column-lanes (pure adds; shift fixed)
#pragma unroll
  for (int s = 1; s < 16; s <<= 1) {
#pragma unroll
    for (int r = 0; r < 8; ++r) {
      zz[r] += __shfl_xor(zz[r], s, 32);
      pp[r] += __shfl_xor(pp[r], s, 32);
      cc[r] += __shfl_xor(cc[r], s, 32);
    }
  }

  if ((lane & 15) == 0) {   // lane 0 -> rows 0..7, lane 16 -> rows 8..15
    float acc = 0.f;
#pragma unroll
    for (int r = 0; r < 8; ++r) {
      float M    = -bias[r];
      float mlpp = pp[r] / (float)cc[r] - M - __logf(zz[r]);
      acc -= mlpp;          // loss = -(T/T_base)*mlpp, ratio == 1
    }
    atomicAdd(out, acc * (1.0f / (float)N_));
  }
}

extern "C" void kernel_launch(void* const* d_in, const int* in_sizes, int n_in,
                              void* d_out, int out_size, void* d_ws, size_t ws_size,
                              hipStream_t stream) {
  const float* features = (const float*)d_in[0];
  const int*   labels   = (const int*)d_in[1];
  float* out = (float*)d_out;

  unsigned short* hi    = (unsigned short*)d_ws;                     // 2 MB
  unsigned short* lo    = hi + (size_t)N_ * D_;                      // 2 MB
  float*          norm2 = (float*)((char*)d_ws + 4u * 1024 * 1024);  // 32 KB
  unsigned int*   maxn  = (unsigned int*)(norm2 + N_);               // 4 B

  init_kernel<<<1, 1, 0, stream>>>(out, maxn);
  convert_kernel<<<(N_ * D_) / 256, 256, 0, stream>>>(features, hi, lo);
  norm_kernel<<<N_ / 256, 256, 0, stream>>>(features, norm2, maxn);
  supcon_kernel<<<128, 128, 0, stream>>>(hi, lo, labels, norm2, maxn, out);
}
